// QWP_56341380989389
// MI455X (gfx1250) — matrix operand order repaired and checked by reading, not compiled, not run on hardware
//
#include <hip/hip_runtime.h>

// ---------------------------------------------------------------------------
// CDNA5 (gfx1250) forward pass for the GNN + edge-transformer reference.
// All large GEMMs and the full 4096x4096 transformer attention run through
// v_wmma_f32_16x16x32_f16 (f16 inputs, f32 accumulate). Fragment loads are
// 2x ds_load_b128 each (A tiles row-major, B tiles column-major); global->LDS
// staging is phase-separated (all loads in flight, then cvt+store) with
// global_load_b128 + ds_store_b64/b32. GEMM K-step is 64 -> 4 WMMAs/barrier.
// ---------------------------------------------------------------------------

typedef __attribute__((ext_vector_type(16))) _Float16 v16h;
typedef __attribute__((ext_vector_type(8)))  _Float16 v8h;
typedef __attribute__((ext_vector_type(4)))  _Float16 v4h;
typedef __attribute__((ext_vector_type(2)))  _Float16 v2h;
typedef __attribute__((ext_vector_type(8)))  float    v8f;

#define HID   128
#define QKV   512
#define CH    128
#define DTR   448
#define D3TR  1344
#define NHEADS 4
#define FHD   112   // transformer head dim
#define FHDP  128   // padded head dim
#define FBQ   64
#define FBK   64

// ---------------- wave/block reduction helpers (wave32) ----------------
__device__ __forceinline__ float wave_sum_f(float v) {
#pragma unroll
  for (int o = 16; o > 0; o >>= 1) v += __shfl_xor(v, o, 32);
  return v;
}

// ordered-uint mapping for float atomic max
__device__ __forceinline__ unsigned f2ord(float f) {
  unsigned u = __float_as_uint(f);
  return (u & 0x80000000u) ? ~u : (u | 0x80000000u);
}
__device__ __forceinline__ float ord2f(unsigned u) {
  unsigned v = (u & 0x80000000u) ? (u & 0x7fffffffu) : ~u;
  return __uint_as_float(v);
}

__device__ __forceinline__ v4h cvt4(float4 f) {
  v4h h; h[0] = (_Float16)f.x; h[1] = (_Float16)f.y;
  h[2] = (_Float16)f.z; h[3] = (_Float16)f.w; return h;
}

// ---------------- WMMA fragment loaders (contiguous k order: k = 16*(lane/16) + s, A and B alike) ----------------
// A fragment (16x32 f16, M x K), tile row-major in LDS, stride `ld` halves
// (ld multiple of 8; base 16B-aligned): one contiguous 32B run per lane (k = kb..kb+15).
__device__ __forceinline__ v16h load_a_frag(const _Float16* base, int ld) {
  const int lane = threadIdx.x & 31;
  const int row  = lane & 15;
  const int kb   = (lane >> 4) << 4;
  const _Float16* p = base + row * ld + kb;
  v8h lo = *(const v8h*)(p);
  v8h hi = *(const v8h*)(p + 8);
  return __builtin_shufflevector(lo, hi, 0, 1, 2, 3, 4, 5, 6, 7,
                                 8, 9, 10, 11, 12, 13, 14, 15);
}

// B fragment (32x16 f16, K x N) from a COLUMN-MAJOR tile buf[n*Kcol + k]:
// one contiguous 32B run per lane (col n = lane%16 + n0, k = kb..kb+15).
__device__ __forceinline__ v16h load_b_frag(const _Float16* buf, int Kcol,
                                            int kOff, int n0) {
  const int lane = threadIdx.x & 31;
  const int n    = lane & 15;
  const int kb   = (lane >> 4) << 4;
  const _Float16* p = buf + (n0 + n) * Kcol + kOff + kb;
  v8h lo = *(const v8h*)(p);
  v8h hi = *(const v8h*)(p + 8);
  return __builtin_shufflevector(lo, hi, 0, 1, 2, 3, 4, 5, 6, 7,
                                 8, 9, 10, 11, 12, 13, 14, 15);
}

__device__ __forceinline__ v8f wmma16(v16h a, v16h b, v8f c) {
  return __builtin_amdgcn_wmma_f32_16x16x32_f16(false, a, false, b, (short)0, c,
                                                false, false);
}

// ---------------------------------------------------------------------------
// Generic GEMM: C[M,N] = act(A[M,K] @ W[K,N] + bias) (+ residual R).
// Block = 256 threads (8 waves), tile 64x64, K-step 64 (4 WMMAs / barrier).
// Phase-separated staging: issue all global loads, then convert + LDS store.
// All K dims used here are multiples of 64. act: 0=none 1=relu 2=gelu(exact)
// ---------------------------------------------------------------------------
__global__ __launch_bounds__(256) void gemm_wmma(
    const float* __restrict__ A, const float* __restrict__ W,
    const float* __restrict__ bias, const float* __restrict__ R,
    float* __restrict__ C, int M, int N, int K, int act)
{
  __shared__ __align__(16) _Float16 As[64 * 72];   // row-major 64x64 (+8 pad)
  __shared__ __align__(16) _Float16 Bs[64 * 64];   // column-major: Bs[n][k]
  const int m0 = blockIdx.x * 64;
  const int n0 = blockIdx.y * 64;
  const int t  = threadIdx.x;
  const int w  = t >> 5, lane = t & 31;
  const int rb  = w >> 1;            // 16-row block (0..3)
  const int cb0 = (w & 1) * 2;       // first 16-col tile (0 or 2)
  v8f acc0, acc1;
#pragma unroll
  for (int i = 0; i < 8; ++i) { acc0[i] = 0.f; acc1[i] = 0.f; }

  for (int k0 = 0; k0 < K; k0 += 64) {
    // ---- load phase: all global loads issued before any use ----
    float4 fa[4];                       // A tile: 64x64 = 1024 float4 chunks
#pragma unroll
    for (int rep = 0; rep < 4; ++rep) {
      const int idx = t + rep * 256;
      const int r = idx >> 4, c4 = (idx & 15) * 4;
      fa[rep] = *(const float4*)(A + (size_t)(m0 + r) * K + (k0 + c4));
    }
    float gb0[8], gb1[8];               // B tile: 64x64 as 32x64 k-pairs
#pragma unroll
    for (int rep = 0; rep < 8; ++rep) {
      const int idx = t + rep * 256;
      const int rp = idx >> 6;          // k-pair index 0..31
      const int c  = idx & 63;          // coalesced over c
      gb0[rep] = W[(size_t)(k0 + 2 * rp) * N + (n0 + c)];
      gb1[rep] = W[(size_t)(k0 + 2 * rp + 1) * N + (n0 + c)];
    }
    if (k0 + 64 < K) {   // prefetch next tiles into L2 (global_prefetch_b8)
      __builtin_prefetch(A + (size_t)(m0 + (t & 63)) * K + (k0 + 64), 0, 3);
      __builtin_prefetch(W + (size_t)(k0 + 64 + (t >> 2)) * N + n0, 0, 3);
    }
    // ---- store phase ----
#pragma unroll
    for (int rep = 0; rep < 4; ++rep) {
      const int idx = t + rep * 256;
      const int r = idx >> 4, c4 = (idx & 15) * 4;
      *(v4h*)(As + r * 72 + c4) = cvt4(fa[rep]);
    }
#pragma unroll
    for (int rep = 0; rep < 8; ++rep) {
      const int idx = t + rep * 256;
      const int rp = idx >> 6;
      const int c  = idx & 63;
      v2h p2; p2[0] = (_Float16)gb0[rep]; p2[1] = (_Float16)gb1[rep];
      *(v2h*)(Bs + c * 64 + 2 * rp) = p2;
    }
    __syncthreads();
#pragma unroll
    for (int ks = 0; ks < 2; ++ks) {
      v16h a  = load_a_frag(&As[(rb * 16) * 72 + ks * 32], 72);
      v16h b0 = load_b_frag(Bs, 64, ks * 32, cb0 * 16);
      v16h b1 = load_b_frag(Bs, 64, ks * 32, (cb0 + 1) * 16);
      acc0 = wmma16(a, b0, acc0);
      acc1 = wmma16(a, b1, acc1);
    }
    __syncthreads();
  }

  const int nn = lane & 15;
  const int mo = (lane >> 4) << 3;
#pragma unroll
  for (int tile = 0; tile < 2; ++tile) {
    v8f acc = tile ? acc1 : acc0;
    const int gn = n0 + (cb0 + tile) * 16 + nn;
    const float bv = bias ? bias[gn] : 0.f;
#pragma unroll
    for (int i = 0; i < 8; ++i) {
      const int gm = m0 + rb * 16 + mo + i;
      float v = acc[i] + bv;
      if (act == 1)      v = fmaxf(v, 0.f);
      else if (act == 2) v = 0.5f * v * (1.f + erff(v * 0.70710678118f));
      if (R) v += R[(size_t)gm * N + gn];
      C[(size_t)gm * N + gn] = v;
    }
  }
}

// ---------------------------------------------------------------------------
// Row LayerNorm: out[r] = (x - mean)/sqrt(var+1e-5)*g + b, 256 threads/row
// ---------------------------------------------------------------------------
__global__ __launch_bounds__(256) void ln_kernel(
    const float* __restrict__ in, float* __restrict__ out,
    const float* __restrict__ g, const float* __restrict__ b, int D)
{
  const int row = blockIdx.x;
  const float* x = in + (size_t)row * D;
  float* y = out + (size_t)row * D;
  __shared__ float red[8];
  const int w = threadIdx.x >> 5, lane = threadIdx.x & 31;

  float s = 0.f;
  for (int j = threadIdx.x; j < D; j += 256) s += x[j];
  s = wave_sum_f(s);
  if (lane == 0) red[w] = s;
  __syncthreads();
  float S = 0.f;
#pragma unroll
  for (int i = 0; i < 8; ++i) S += red[i];
  const float mean = S / (float)D;
  __syncthreads();

  float s2 = 0.f;
  for (int j = threadIdx.x; j < D; j += 256) { float d = x[j] - mean; s2 += d * d; }
  s2 = wave_sum_f(s2);
  if (lane == 0) red[w] = s2;
  __syncthreads();
  float V = 0.f;
#pragma unroll
  for (int i = 0; i < 8; ++i) V += red[i];
  const float inv = rsqrtf(V / (float)D + 1e-5f);

  for (int j = threadIdx.x; j < D; j += 256)
    y[j] = (x[j] - mean) * inv * g[j] + b[j];
}

__global__ void fill_kernel(unsigned* __restrict__ p, unsigned v, int n) {
  int i = blockIdx.x * blockDim.x + threadIdx.x;
  if (i < n) p[i] = v;
}

// ---------------------------------------------------------------------------
// Per-node feature embedding: te/ce/de/pe MLPs + LNs, stab_emb LN, *syn.
// Writes embeds[N][256]. 128 threads/node; waves handle the four 32-d LNs.
// ---------------------------------------------------------------------------
__global__ __launch_bounds__(128) void embed_kernel(
    const float* __restrict__ x, const int* __restrict__ syn,
    const float* te_w, const float* te_b, const float* te_g, const float* te_bb,
    const float* c1_w, const float* c1_b, const float* c2_w, const float* c2_b,
    const float* c_g, const float* c_bb,
    const float* d1_w, const float* d1_b, const float* d2_w, const float* d2_b,
    const float* d_g, const float* d_bb,
    const float* p1_w, const float* p1_b, const float* p2_w, const float* p2_b,
    const float* p_g, const float* p_bb,
    const float* stab, const float* e_g, const float* e_bb,
    float* __restrict__ embeds)
{
  const int n = blockIdx.x, t = threadIdx.x;
  const int w = t >> 5, lane = t & 31;
  __shared__ float xs[16];
  __shared__ float h1c[128], h1d[128], h1p[128];
  __shared__ float red[8];
  if (t < 13) xs[t] = x[n * 13 + t];
  __syncthreads();
  {
    float a = xs[2] * c1_w[t] + xs[3] * c1_w[128 + t] + c1_b[t];
    h1c[t] = fmaxf(a, 0.f);
    a = xs[4] * d1_w[t] + d1_b[t];
    h1d[t] = fmaxf(a, 0.f);
    a = p1_b[t];
#pragma unroll
    for (int i = 0; i < 8; ++i) a += xs[5 + i] * p1_w[i * 128 + t];
    h1p[t] = fmaxf(a, 0.f);
  }
  __syncthreads();
  const float sf = (syn[n] != 0) ? 1.f : -1.f;

  float val;
  const float *g = te_g, *bb = te_bb;
  if (w == 0) {
    val = xs[0] * te_w[lane] + xs[1] * te_w[32 + lane] + te_b[lane];
  } else if (w == 1) {
    val = c2_b[lane];
    for (int i = 0; i < 128; ++i) val += h1c[i] * c2_w[i * 32 + lane];
    g = c_g; bb = c_bb;
  } else if (w == 2) {
    val = d2_b[lane];
    for (int i = 0; i < 128; ++i) val += h1d[i] * d2_w[i * 32 + lane];
    g = d_g; bb = d_bb;
  } else {
    val = p2_b[lane];
    for (int i = 0; i < 128; ++i) val += h1p[i] * p2_w[i * 32 + lane];
    g = p_g; bb = p_bb;
  }
  const float mean = wave_sum_f(val) * (1.f / 32.f);
  const float dv   = val - mean;
  const float var  = wave_sum_f(dv * dv) * (1.f / 32.f);
  embeds[(size_t)n * 256 + w * 32 + lane] =
      (dv * rsqrtf(var + 1e-5f) * g[lane] + bb[lane]) * sf;

  // learned stabilizer embedding LN (128-wide, cross-wave)
  const float lv = stab[(size_t)n * 128 + t];
  float s = wave_sum_f(lv);
  if (lane == 0) red[w] = s;
  __syncthreads();
  const float m2 = (red[0] + red[1] + red[2] + red[3]) * (1.f / 128.f);
  __syncthreads();
  const float d2 = lv - m2;
  s = wave_sum_f(d2 * d2);
  if (lane == 0) red[w] = s;
  __syncthreads();
  const float inv = rsqrtf((red[0] + red[1] + red[2] + red[3]) * (1.f / 128.f) + 1e-5f);
  embeds[(size_t)n * 256 + 128 + t] = (d2 * inv * e_g[t] + e_bb[t]) * sf;
}

// ---------------------------------------------------------------------------
// Per-edge features: dist_e = LN128(distance_emb[id]); rem = LN64(relu(...)) x2
// ---------------------------------------------------------------------------
__global__ __launch_bounds__(128) void edgefeat_kernel(
    const float* __restrict__ ea, const float* demb,
    const float* dn_g, const float* dn_b,
    const float* r1_w, const float* r1_b, const float* l1_g, const float* l1_b,
    const float* r2_w, const float* r2_b, const float* l2_g, const float* l2_b,
    float* __restrict__ dist_e, float* __restrict__ remf)
{
  const int e = blockIdx.x, t = threadIdx.x;
  const int w = t >> 5, lane = t & 31;
  __shared__ float red[4];
  __shared__ float r[64];

  int id = (int)ea[e * 4 + 0];
  if (id > 32) id = 32;
  const float v = demb[(size_t)id * 128 + t];
  float s = wave_sum_f(v);
  if (lane == 0) red[w] = s;
  __syncthreads();
  float mean = (red[0] + red[1] + red[2] + red[3]) * (1.f / 128.f);
  __syncthreads();
  const float d = v - mean;
  s = wave_sum_f(d * d);
  if (lane == 0) red[w] = s;
  __syncthreads();
  float inv = rsqrtf((red[0] + red[1] + red[2] + red[3]) * (1.f / 128.f) + 1e-5f);
  dist_e[(size_t)e * 128 + t] = d * inv * dn_g[t] + dn_b[t];
  __syncthreads();

  const float ea1 = ea[e * 4 + 1], ea2 = ea[e * 4 + 2], ea3 = ea[e * 4 + 3];
  if (t < 64)
    r[t] = fmaxf(r1_b[t] + ea1 * r1_w[t] + ea2 * r1_w[64 + t] + ea3 * r1_w[128 + t], 0.f);
  __syncthreads();

  // LN64 #1
  float v2 = (t < 64) ? r[t] : 0.f;
  s = wave_sum_f(v2);
  if (lane == 0) red[w] = s;
  __syncthreads();
  mean = (red[0] + red[1]) * (1.f / 64.f);
  __syncthreads();
  float dd = v2 - mean;
  s = wave_sum_f((t < 64) ? dd * dd : 0.f);
  if (lane == 0) red[w] = s;
  __syncthreads();
  inv = rsqrtf((red[0] + red[1]) * (1.f / 64.f) + 1e-5f);
  __syncthreads();
  if (t < 64) r[t] = dd * inv * l1_g[t] + l1_b[t];
  __syncthreads();

  // second linear + relu
  float h2 = 0.f;
  if (t < 64) {
    float a = r2_b[t];
    for (int i = 0; i < 64; ++i) a += r[i] * r2_w[i * 64 + t];
    h2 = fmaxf(a, 0.f);
  }
  __syncthreads();
  // LN64 #2
  s = wave_sum_f(h2);
  if (lane == 0) red[w] = s;
  __syncthreads();
  mean = (red[0] + red[1]) * (1.f / 64.f);
  __syncthreads();
  dd = h2 - mean;
  s = wave_sum_f((t < 64) ? dd * dd : 0.f);
  if (lane == 0) red[w] = s;
  __syncthreads();
  inv = rsqrtf((red[0] + red[1]) * (1.f / 64.f) + 1e-5f);
  if (t < 64) remf[(size_t)e * 64 + t] = dd * inv * l2_g[t] + l2_b[t];
}

// ---------------------------------------------------------------------------
// GNN per-edge attention: logits + scatter max / sum / weighted aggregate
// ---------------------------------------------------------------------------
__global__ __launch_bounds__(128) void alpha_kernel(
    const int* __restrict__ ei, int E,
    const float* __restrict__ q, const float* __restrict__ k,
    float* __restrict__ alpha, unsigned* __restrict__ amax)
{
  const int e = blockIdx.x;
  const int t = threadIdx.x, hh = t >> 5, lane = t & 31;
  const int s = ei[e], g = ei[E + e];
  float sum = 0.f;
  for (int c = lane; c < CH; c += 32)
    sum += q[(size_t)g * QKV + hh * CH + c] * k[(size_t)s * QKV + hh * CH + c];
  sum = wave_sum_f(sum);
  if (lane == 0) {
    const float al = sum * 0.08838834764831845f;   // 1/sqrt(128)
    alpha[e * 4 + hh] = al;
    atomicMax(&amax[g * 4 + hh], f2ord(al));
  }
}

__global__ void ex_kernel(const int* __restrict__ ei, int E,
    const float* __restrict__ alpha, const unsigned* __restrict__ amax,
    float* __restrict__ exb, float* __restrict__ denom)
{
  const int idx = blockIdx.x * blockDim.x + threadIdx.x;
  if (idx >= E * 4) return;
  const int e = idx >> 2, hh = idx & 3;
  const int g = ei[E + e];
  const float ex = expf(alpha[idx] - ord2f(amax[g * 4 + hh]));
  exb[idx] = ex;
  atomicAdd(&denom[g * 4 + hh], ex);
}

__global__ __launch_bounds__(128) void aggr_kernel(const int* __restrict__ ei, int E,
    const float* __restrict__ vbuf, const float* __restrict__ exb,
    const float* __restrict__ denom, float* __restrict__ agg)
{
  const int e = blockIdx.x;
  const int s = ei[e], g = ei[E + e];
#pragma unroll
  for (int rep = 0; rep < 4; ++rep) {
    const int j = threadIdx.x + rep * 128;
    const int hh = j >> 7;
    const float wgt = exb[e * 4 + hh] / (denom[g * 4 + hh] + 1e-16f);
    atomicAdd(&agg[(size_t)g * QKV + j], vbuf[(size_t)s * QKV + j] * wgt);
  }
}

// mean-over-heads, beta gate, residual
__global__ __launch_bounds__(128) void combine_kernel(const float* __restrict__ agg,
    const float* __restrict__ xr, const float* __restrict__ bw,
    float* __restrict__ h)
{
  const int n = blockIdx.x, t = threadIdx.x;
  const int w = t >> 5, lane = t & 31;
  __shared__ float red[4];
  const float o = 0.25f * (agg[(size_t)n * QKV + t] + agg[(size_t)n * QKV + 128 + t] +
                           agg[(size_t)n * QKV + 256 + t] + agg[(size_t)n * QKV + 384 + t]);
  const float xrv = xr[(size_t)n * HID + t];
  float contrib = o * bw[t] + xrv * bw[HID + t] + (o - xrv) * bw[2 * HID + t];
  contrib = wave_sum_f(contrib);
  if (lane == 0) red[w] = contrib;
  __syncthreads();
  const float S = red[0] + red[1] + red[2] + red[3];
  const float beta = 1.f / (1.f + expf(-S));
  h[(size_t)n * HID + t] = beta * xrv + (1.f - beta) * o + h[(size_t)n * HID + t];
}

__global__ __launch_bounds__(128) void concat_kernel(const int* __restrict__ ei, int E,
    const float* __restrict__ h, const float* __restrict__ dist_e,
    const float* __restrict__ rem, float* __restrict__ out)
{
  const int e = blockIdx.x, t = threadIdx.x;
  const int s = ei[e], g = ei[E + e];
  out[(size_t)e * DTR + t]       = h[(size_t)s * HID + t];
  out[(size_t)e * DTR + 128 + t] = h[(size_t)g * HID + t];
  out[(size_t)e * DTR + 256 + t] = dist_e[(size_t)e * 128 + t];
  if (t < 64) out[(size_t)e * DTR + 384 + t] = rem[(size_t)e * 64 + t];
}

// ---------------------------------------------------------------------------
// Flash attention over all E edges, per head. hd=112 padded to 128.
// Block = 256 threads (8 waves). Q-block 64 rows; stream 64-key blocks.
// S = Q*K^T and O += P*V through v_wmma_f32_16x16x32_f16.
// ---------------------------------------------------------------------------
__global__ __launch_bounds__(256) void flash_attn(const float* __restrict__ qkv,
                                                  float* __restrict__ o, int E)
{
  const int head = blockIdx.x;
  const int qb   = blockIdx.y;
  const int t = threadIdx.x, w = t >> 5, lane = t & 31;
  const int rb    = w >> 1;      // 16-row block of this wave
  const int chalf = w & 1;       // 64-col half of this wave

  __shared__ __align__(16) _Float16 ktc[FBK * FHDP];  // [key][dim] (B col-major)
  __shared__ __align__(16) _Float16 vtc[FHDP * FBK];  // [dim][key] (B col-major)
  __shared__ __align__(16) float    sb[FBQ * FBK];    // scores (aliased: f16 Q)
  __shared__ __align__(16) _Float16 pb[FBQ * FBK];    // probabilities (A row-major)
  __shared__ float rowM[FBQ], rowL[FBQ], rowS[FBQ];

  _Float16* qt = (_Float16*)sb;         // 64*128 f16 == 16KB == sb
  const int qrow0 = qb * FBQ;
  // Q staging: 64 rows x 32 float4 chunks (28 real + 4 pad), 2 chunks / wait
#pragma unroll
  for (int rep2 = 0; rep2 < 4; ++rep2) {
    float4 f[2] = {};
    int rr[2], cc[2];
#pragma unroll
    for (int j = 0; j < 2; ++j) {
      const int idx = t + (rep2 * 2 + j) * 256;
      rr[j] = idx >> 5; cc[j] = (idx & 31) * 4;
      if (cc[j] < FHD)
        f[j] = *(const float4*)(qkv + (size_t)(qrow0 + rr[j]) * D3TR + head * FHD + cc[j]);
    }
#pragma unroll
    for (int j = 0; j < 2; ++j)
      *(v4h*)(qt + rr[j] * FHDP + cc[j]) = cvt4(f[j]);
  }
  if (t < FBQ) { rowM[t] = -1e30f; rowL[t] = 0.f; }
  __syncthreads();

  v16h qf[4];
#pragma unroll
  for (int ks = 0; ks < 4; ++ks)
    qf[ks] = load_a_frag(&qt[(rb * 16) * FHDP + ks * 32], FHDP);
  __syncthreads();   // qt (sb) free for reuse

  v8f oacc[4];
#pragma unroll
  for (int ct = 0; ct < 4; ++ct)
#pragma unroll
    for (int i = 0; i < 8; ++i) oacc[ct][i] = 0.f;

  const float sc = 0.09449111825230681f;   // 1/sqrt(112)
  const int nblocks = E / FBK;
  for (int kb = 0; kb < nblocks; ++kb) {
    const int krow0 = kb * FBK;
    // K/V staging: 64 keys x 32 float4 chunks each, 2 chunks (4 loads) / wait
#pragma unroll
    for (int rep2 = 0; rep2 < 4; ++rep2) {
      float4 fk[2] = {}, fv[2] = {};
      int nn_[2], cc[2];
#pragma unroll
      for (int j = 0; j < 2; ++j) {
        const int idx = t + (rep2 * 2 + j) * 256;
        nn_[j] = idx >> 5; cc[j] = (idx & 31) * 4;
        if (cc[j] < FHD) {
          const float* rowp =
              qkv + (size_t)(krow0 + nn_[j]) * D3TR + head * FHD + cc[j];
          fk[j] = *(const float4*)(rowp + DTR);
          fv[j] = *(const float4*)(rowp + 2 * DTR);
        }
      }
#pragma unroll
      for (int j = 0; j < 2; ++j) {
        *(v4h*)(ktc + nn_[j] * FHDP + cc[j]) = cvt4(fk[j]);
        vtc[(cc[j] + 0) * FBK + nn_[j]] = (_Float16)fv[j].x;
        vtc[(cc[j] + 1) * FBK + nn_[j]] = (_Float16)fv[j].y;
        vtc[(cc[j] + 2) * FBK + nn_[j]] = (_Float16)fv[j].z;
        vtc[(cc[j] + 3) * FBK + nn_[j]] = (_Float16)fv[j].w;
      }
    }
    if (kb + 1 < nblocks) {   // prefetch next key block (global_prefetch_b8)
      __builtin_prefetch(qkv + (size_t)(krow0 + FBK + (t >> 2)) * D3TR + DTR + head * FHD, 0, 3);
      __builtin_prefetch(qkv + (size_t)(krow0 + FBK + (t >> 2)) * D3TR + 2 * DTR + head * FHD, 0, 3);
    }
    __syncthreads();

    // S tile: rows rb*16, cols chalf*32 + {0,16}; K over padded head dim
    v8f s0, s1;
#pragma unroll
    for (int i = 0; i < 8; ++i) { s0[i] = 0.f; s1[i] = 0.f; }
#pragma unroll
    for (int ks = 0; ks < 4; ++ks) {
      v16h b0 = load_b_frag(ktc, FHDP, ks * 32, chalf * 32);
      v16h b1 = load_b_frag(ktc, FHDP, ks * 32, chalf * 32 + 16);
      s0 = wmma16(qf[ks], b0, s0);
      s1 = wmma16(qf[ks], b1, s1);
    }
    {
      const int nn = lane & 15, mo = (lane >> 4) << 3;
#pragma unroll
      for (int i = 0; i < 8; ++i) {
        const int r = rb * 16 + mo + i;
        sb[r * FBK + chalf * 32 + nn]      = s0[i] * sc;
        sb[r * FBK + chalf * 32 + 16 + nn] = s1[i] * sc;
      }
    }
    __syncthreads();

    // online softmax update, one thread per query row
    if (t < FBQ) {
      float m = rowM[t];
      for (int c = 0; c < FBK; ++c) m = fmaxf(m, sb[t * FBK + c]);
      const float scale = expf(rowM[t] - m);
      float l = rowL[t] * scale;
      for (int c = 0; c < FBK; ++c) {
        const float p = expf(sb[t * FBK + c] - m);
        pb[t * FBK + c] = (_Float16)p;
        l += p;
      }
      rowM[t] = m; rowL[t] = l; rowS[t] = scale;
    }
    __syncthreads();

    // rescale O, then O += P * V (K over the 64 keys)
    {
      const int mo = (lane >> 4) << 3;
#pragma unroll
      for (int ct = 0; ct < 4; ++ct)
#pragma unroll
        for (int i = 0; i < 8; ++i) oacc[ct][i] *= rowS[rb * 16 + mo + i];
#pragma unroll
      for (int ks = 0; ks < 2; ++ks) {
        v16h a = load_a_frag(&pb[(rb * 16) * FBK + ks * 32], FBK);
#pragma unroll
        for (int ct = 0; ct < 4; ++ct) {
          v16h b = load_b_frag(vtc, FBK, ks * 32, chalf * 64 + ct * 16);
          oacc[ct] = wmma16(a, b, oacc[ct]);
        }
      }
    }
    __syncthreads();
  }

  // write O / l
  const int nn = lane & 15, mo = (lane >> 4) << 3;
#pragma unroll
  for (int ct = 0; ct < 4; ++ct) {
    const int col = chalf * 64 + ct * 16 + nn;
    if (col < FHD) {
#pragma unroll
      for (int i = 0; i < 8; ++i) {
        const int r = rb * 16 + mo + i;
        o[(size_t)(qrow0 + r) * DTR + head * FHD + col] = oacc[ct][i] / rowL[r];
      }
    }
  }
}

// final projection 448 -> 1
__global__ __launch_bounds__(128) void outdot_kernel(const float* __restrict__ e,
    const float* __restrict__ ow, const float* __restrict__ ob,
    float* __restrict__ out)
{
  const int r = blockIdx.x, t = threadIdx.x;
  const int w = t >> 5, lane = t & 31;
  __shared__ float red[4];
  float s = e[(size_t)r * DTR + t] * ow[t] +
            e[(size_t)r * DTR + 128 + t] * ow[128 + t] +
            e[(size_t)r * DTR + 256 + t] * ow[256 + t];
  if (t < 64) s += e[(size_t)r * DTR + 384 + t] * ow[384 + t];
  s = wave_sum_f(s);
  if (lane == 0) red[w] = s;
  __syncthreads();
  if (t == 0) out[r] = red[0] + red[1] + red[2] + red[3] + ob[0];
}

// ---------------------------------------------------------------------------
// Host orchestration. Parameter leaf order assumed: top-level inputs in
// insertion order (x, edge_index, edge_attr, syndrome), then the params
// pytree flattened with sorted dict keys (jax convention), lists in order.
// ---------------------------------------------------------------------------
extern "C" void kernel_launch(void* const* d_in, const int* in_sizes, int n_in,
                              void* d_out, int out_size, void* d_ws, size_t ws_size,
                              hipStream_t stream)
{
  const float* x   = (const float*)d_in[0];
  const int*   ei  = (const int*)d_in[1];
  const float* ea  = (const float*)d_in[2];
  const int*   syn = (const int*)d_in[3];
  auto P = [&](int i) { return (const float*)d_in[i]; };

  const int N = in_sizes[0] / 13;
  const int E = in_sizes[2] / 4;

  char* wsb = (char*)d_ws;
  size_t off = 0;
  auto alloc = [&](size_t bytes) -> float* {
    float* p = (float*)(wsb + off);
    off += (bytes + 255) & ~(size_t)255;
    return p;
  };
  float* embeds = alloc((size_t)N * 256 * 4);
  float* h      = alloc((size_t)N * HID * 4);
  float* xn     = alloc((size_t)N * HID * 4);
  float* tmp128 = alloc((size_t)N * HID * 4);
  float* qb     = alloc((size_t)N * QKV * 4);
  float* kb     = alloc((size_t)N * QKV * 4);
  float* vb     = alloc((size_t)N * QKV * 4);
  float* xr     = alloc((size_t)N * HID * 4);
  float* agg    = alloc((size_t)N * QKV * 4);
  float* mid    = alloc((size_t)N * 512 * 4);
  float* alphab = alloc((size_t)E * 4 * 4);
  float* exb    = alloc((size_t)E * 4 * 4);
  unsigned* amax = (unsigned*)alloc((size_t)N * 4 * 4);
  float* denom  = alloc((size_t)N * 4 * 4);
  float* dist_e = alloc((size_t)E * 128 * 4);
  float* remb   = alloc((size_t)E * 64 * 4);
  float* ebuf   = alloc((size_t)E * DTR * 4);
  float* etmp   = alloc((size_t)E * DTR * 4);
  float* qkvb   = qb;    // q/k/v region (3x N*512) reused for E*1344 qkv
  float* obuf   = agg;   // agg region reused for attention output (E*448)

  // node embedding -> pre_proj GEMM -> pre_norm
  embed_kernel<<<N, 128, 0, stream>>>(x, syn,
      P(137), P(136), P(139), P(138),                 // type_emb w/b, type_norm g/b
      P(5), P(4), P(7), P(6), P(9), P(8),             // coords1/2, coords_norm
      P(11), P(10), P(13), P(12), P(15), P(14),       // dist1/2, dist_norm
      P(92), P(91), P(94), P(93), P(96), P(95),       // pe1/2, pe_norm
      P(111), P(20), P(19),                           // stab_emb, emb_norm g/b
      embeds);
  gemm_wmma<<<dim3(N / 64, HID / 64), 256, 0, stream>>>(
      embeds, P(100), P(99), nullptr, tmp128, N, HID, 256, 0);
  ln_kernel<<<N, 256, 0, stream>>>(tmp128, h, P(98), P(97), HID);

  // edge features (independent of h)
  edgefeat_kernel<<<E, 128, 0, stream>>>(ea, P(16), P(18), P(17),
      P(104), P(103), P(108), P(107), P(106), P(105), P(110), P(109),
      dist_e, remb);

  // GNN layers
  for (int l = 0; l < 4; ++l) {
    const int b0 = 21 + 17 * l;
    ln_kernel<<<N, 256, 0, stream>>>(h, xn, P(b0 + 8), P(b0 + 7), HID);   // norm1
    gemm_wmma<<<dim3(N / 64, QKV / 64), 256, 0, stream>>>(xn, P(b0 + 12), P(b0 + 11), nullptr, qb, N, QKV, HID, 0);
    gemm_wmma<<<dim3(N / 64, QKV / 64), 256, 0, stream>>>(xn, P(b0 + 6),  P(b0 + 5),  nullptr, kb, N, QKV, HID, 0);
    gemm_wmma<<<dim3(N / 64, QKV / 64), 256, 0, stream>>>(xn, P(b0 + 16), P(b0 + 15), nullptr, vb, N, QKV, HID, 0);
    gemm_wmma<<<dim3(N / 64, HID / 64), 256, 0, stream>>>(xn, P(b0 + 14), P(b0 + 13), nullptr, xr, N, HID, HID, 0);
    fill_kernel<<<(N * 4 + 255) / 256, 256, 0, stream>>>(amax, 0x007FFFFFu, N * 4); // ordered(-inf)
    fill_kernel<<<(N * 4 + 255) / 256, 256, 0, stream>>>((unsigned*)denom, 0u, N * 4);
    fill_kernel<<<(N * QKV + 255) / 256, 256, 0, stream>>>((unsigned*)agg, 0u, N * QKV);
    alpha_kernel<<<E, 128, 0, stream>>>(ei, E, qb, kb, alphab, amax);
    ex_kernel<<<(E * 4 + 255) / 256, 256, 0, stream>>>(ei, E, alphab, amax, exb, denom);
    aggr_kernel<<<E, 128, 0, stream>>>(ei, E, vb, exb, denom, agg);
    combine_kernel<<<N, 128, 0, stream>>>(agg, xr, P(b0), h);
    ln_kernel<<<N, 256, 0, stream>>>(h, xn, P(b0 + 10), P(b0 + 9), HID);  // norm2
    gemm_wmma<<<dim3(N / 64, 512 / 64), 256, 0, stream>>>(xn, P(b0 + 2), P(b0 + 1), nullptr, mid, N, 512, HID, 2);
    gemm_wmma<<<dim3(N / 64, HID / 64), 256, 0, stream>>>(mid, P(b0 + 4), P(b0 + 3), h, h, N, HID, 512, 0);
  }

  // edge sequence -> transformer
  concat_kernel<<<E, 128, 0, stream>>>(ei, E, h, dist_e, remb, etmp);
  ln_kernel<<<E, 256, 0, stream>>>(etmp, ebuf, P(102), P(101), DTR);      // pred_norm
  for (int ti = 0; ti < 2; ++ti) {
    const int t0 = 112 + 12 * ti;
    gemm_wmma<<<dim3(E / 64, D3TR / 64), 256, 0, stream>>>(ebuf, P(t0 + 1), P(t0), nullptr, qkvb, E, D3TR, DTR, 0);
    flash_attn<<<dim3(NHEADS, E / 64), 256, 0, stream>>>(qkvb, obuf, E);
    gemm_wmma<<<dim3(E / 64, DTR / 64), 256, 0, stream>>>(obuf, P(t0 + 11), P(t0 + 10), ebuf, etmp, E, DTR, DTR, 0);
    ln_kernel<<<E, 256, 0, stream>>>(etmp, ebuf, P(t0 + 7), P(t0 + 6), DTR);   // n1
    gemm_wmma<<<dim3(E / 64, 512 / 64), 256, 0, stream>>>(ebuf, P(t0 + 3), P(t0 + 2), nullptr, mid, E, 512, DTR, 1);
    gemm_wmma<<<dim3(E / 64, DTR / 64), 256, 0, stream>>>(mid, P(t0 + 5), P(t0 + 4), ebuf, etmp, E, DTR, 512, 0);
    ln_kernel<<<E, 256, 0, stream>>>(etmp, ebuf, P(t0 + 9), P(t0 + 8), DTR);   // n2
  }
  outdot_kernel<<<E, 128, 0, stream>>>(ebuf, P(90), P(89), (float*)d_out);
  (void)n_in; (void)out_size; (void)ws_size;
}
